// Actor_55216099557844
// MI455X (gfx1250) — matrix operand order repaired and checked by reading, not compiled, not run on hardware
//
#include <hip/hip_runtime.h>

#define VOCAB 32000
#define EMB   128
#define HID   512
#define BATCH 32
#define SEQ   128
#define UNIFORM_LP (-10.3734912f)   // -log(32000)

typedef __attribute__((ext_vector_type(16))) _Float16 v16h;
typedef __attribute__((ext_vector_type(8)))  float    v8f;

// ---- float <-> total-order unsigned key (unsigned compare == float compare) ----
__device__ __forceinline__ unsigned fkey(float f) {
  unsigned b = __float_as_uint(f);
  return (b & 0x80000000u) ? ~b : (b | 0x80000000u);
}
__device__ __forceinline__ float fval(unsigned k) {
  unsigned b = (k & 0x80000000u) ? (k & 0x7FFFFFFFu) : ~k;
  return __uint_as_float(b);
}
__device__ __forceinline__ unsigned long long shflxor64(unsigned long long v, int m) {
  unsigned lo = (unsigned)(v & 0xFFFFFFFFull);
  unsigned hi = (unsigned)(v >> 32);
  lo = __shfl_xor(lo, m, 32);
  hi = __shfl_xor(hi, m, 32);
  return (((unsigned long long)hi) << 32) | (unsigned long long)lo;
}

// ---------------- one-time conversion / init ----------------
__global__ void cvt_f16_kernel(_Float16* __restrict__ dst, const float* __restrict__ src, int n) {
  int i = blockIdx.x * 256 + threadIdx.x;
  if (i < n) dst[i] = (_Float16)src[i];
}

__global__ void init_state_kernel(float* __restrict__ hf32, _Float16* __restrict__ hf16,
                                  _Float16* __restrict__ xf16, const float* __restrict__ emb) {
  int i = blockIdx.x * 256 + threadIdx.x;
  if (i < BATCH * HID) { hf32[i] = 0.0f; hf16[i] = (_Float16)0.0f; }
  if (i < BATCH * EMB) xf16[i] = (_Float16)emb[i & (EMB - 1)];   // embedding[0] broadcast
}

// ---------------- WMMA tile body: one wave owns one 16-col tile, full M=32 ----------------
// Two v_wmma per k-block share a single B register load (halves B/L2 traffic).
__device__ __forceinline__ void gemm_m32_tile(const _Float16* __restrict__ A,  // [32, K]
                                              const _Float16* __restrict__ W,  // [N, K]
                                              const float* __restrict__ bias,  // [N] | null
                                              float* __restrict__ C,           // [32, N]
                                              unsigned* __restrict__ rowmax,   // [32] | null
                                              int N, int K, int ntile, int lane) {
  const int nl = lane & 15;
  const int h  = lane >> 4;
  const int wrow = ntile * 16 + nl;          // output column / weight row
  float bv = bias ? bias[wrow] : 0.0f;
  v8f acc0, acc1;
#pragma unroll
  for (int r = 0; r < 8; ++r) { acc0[r] = bv; acc1[r] = bv; }

  const _Float16* Ar0 = A + (size_t)nl * K;         // batch rows 0..15
  const _Float16* Ar1 = A + (size_t)(16 + nl) * K;  // batch rows 16..31
  const _Float16* Wr  = W + (size_t)wrow * K + h * 16;
  const int nkb = K >> 5;
  for (int kb = 0; kb < nkb; ++kb) {
    v16h a0, a1;
#pragma unroll
    for (int v = 0; v < 8; ++v) {
      int k = kb * 32 + (h << 4) + (v << 1);
      a0[2 * v] = Ar0[k]; a0[2 * v + 1] = Ar0[k + 1];
      a1[2 * v] = Ar1[k]; a1[2 * v + 1] = Ar1[k + 1];
    }
    v16h b = *(const v16h*)(Wr + kb * 32);  // 32B aligned chunk of weight row, loaded once
    acc0 = __builtin_amdgcn_wmma_f32_16x16x32_f16(false, a0, false, b, (short)0, acc0,
                                                  false, false);
    acc1 = __builtin_amdgcn_wmma_f32_16x16x32_f16(false, a1, false, b, (short)0, acc1,
                                                  false, false);
  }
#pragma unroll
  for (int r = 0; r < 8; ++r) {             // D layout: m = r + 8*h, n = lane%16
    C[(size_t)(h * 8 + r) * N + wrow]        = acc0[r];
    C[(size_t)(16 + h * 8 + r) * N + wrow]   = acc1[r];
  }
  if (rowmax) {
#pragma unroll
    for (int r = 0; r < 8; ++r) {
      float m0 = acc0[r], m1 = acc1[r];
#pragma unroll
      for (int msk = 1; msk <= 8; msk <<= 1) {
        m0 = fmaxf(m0, __shfl_xor(m0, msk, 32));
        m1 = fmaxf(m1, __shfl_xor(m1, msk, 32));
      }
      if (nl == 0) {
        atomicMax(rowmax + (h * 8 + r), fkey(m0));
        atomicMax(rowmax + (16 + h * 8 + r), fkey(m1));
      }
    }
  }
}

// logits = h @ w_dist^T + b_dist, fused per-row max; 2000 waves
__global__ void logits_gemm_kernel(const _Float16* __restrict__ A, const _Float16* __restrict__ W,
                                   const float* __restrict__ bias, float* __restrict__ C,
                                   unsigned* __restrict__ rowmax) {
  const int wave = (int)((blockIdx.x * blockDim.x + threadIdx.x) >> 5);
  if (wave >= VOCAB / 16) return;           // wave-uniform exit, EXEC stays all-ones
  gemm_m32_tile(A, W, bias, C, rowmax, VOCAB, HID, wave, threadIdx.x & 31);
}

// both gate GEMMs in one launch: waves 0..95 -> Gi (K=128), waves 96..191 -> Gh (K=512)
__global__ void gates_gemm_kernel(const _Float16* __restrict__ x, const _Float16* __restrict__ Wx,
                                  const float* __restrict__ bx, float* __restrict__ Gi,
                                  const _Float16* __restrict__ hh, const _Float16* __restrict__ Wh,
                                  const float* __restrict__ bh, float* __restrict__ Gh) {
  const int wave = (int)((blockIdx.x * blockDim.x + threadIdx.x) >> 5);
  if (wave >= 2 * (3 * HID / 16)) return;
  const int sel   = wave >= (3 * HID / 16);
  const int ntile = sel ? wave - (3 * HID / 16) : wave;
  if (sel)
    gemm_m32_tile(hh, Wh, bh, Gh, nullptr, 3 * HID, HID, ntile, threadIdx.x & 31);
  else
    gemm_m32_tile(x, Wx, bx, Gi, nullptr, 3 * HID, EMB, ntile, threadIdx.x & 31);
}

// ---------------- GRU pointwise + per-step scratch reset ----------------
__global__ void gru_pointwise_kernel(const float* __restrict__ Gi, const float* __restrict__ Gh,
                                     float* __restrict__ hf32, _Float16* __restrict__ hf16,
                                     unsigned* __restrict__ rowmax, float* __restrict__ rowsum,
                                     unsigned long long* __restrict__ argkey,
                                     unsigned* __restrict__ done_ctr) {
  int idx = blockIdx.x * 256 + threadIdx.x;
  if (idx < BATCH) { rowmax[idx] = 0u; rowsum[idx] = 0.0f; argkey[idx] = 0ull; }
  if (idx == 0) *done_ctr = 0u;
  if (idx >= BATCH * HID) return;
  int b = idx >> 9, j = idx & (HID - 1);
  const float* gi = Gi + b * 3 * HID;
  const float* gh = Gh + b * 3 * HID;
  float ir = gi[j], iz = gi[HID + j], in = gi[2 * HID + j];
  float hr = gh[j], hz = gh[HID + j], hn = gh[2 * HID + j];
  float r = 1.0f / (1.0f + __expf(-(ir + hr)));
  float z = 1.0f / (1.0f + __expf(-(iz + hz)));
  float n = tanhf(in + r * hn);
  float hnew = (1.0f - z) * n + z * hf32[idx];
  hf32[idx] = hnew;
  hf16[idx] = (_Float16)hnew;
}

// ---------------- per-row sum of exp(logit - max) ----------------
__global__ void row_sumexp_kernel(const float* __restrict__ logits,
                                  const unsigned* __restrict__ rowmax,
                                  float* __restrict__ rowsum) {
  __shared__ float sh[256];
  int b = blockIdx.x >> 4;       // 32 rows
  int chunk = blockIdx.x & 15;   // 16 chunks of 2000 columns
  float mx = fval(rowmax[b]);
  float s = 0.0f;
  for (int i = threadIdx.x; i < 2000; i += 256)
    s += __expf(logits[(size_t)b * VOCAB + chunk * 2000 + i] - mx);
  sh[threadIdx.x] = s;
  __syncthreads();
  for (int off = 128; off > 0; off >>= 1) {
    if (threadIdx.x < off) sh[threadIdx.x] += sh[threadIdx.x + off];
    __syncthreads();
  }
  if (threadIdx.x == 0) atomicAdd(rowsum + b, sh[0]);
}

// ---- logp/prob/mean_probs + Gumbel argmax + (last block) sampling & gather ----
__global__ void finalize_kernel(const float* __restrict__ logits,
                                const unsigned* __restrict__ rowmax,
                                const float* __restrict__ rowsum,
                                const float* __restrict__ u_eps,     // [T,B]
                                const float* __restrict__ u_gumbel,  // [T,B,V]
                                const float* __restrict__ emb,
                                float* __restrict__ d_logp, float* __restrict__ d_prob,
                                float* __restrict__ d_mean,
                                float* __restrict__ out_tok, float* __restrict__ out_corr,
                                _Float16* __restrict__ xf16,
                                unsigned long long* __restrict__ argkey,
                                unsigned* __restrict__ done_ctr, int t) {
  __shared__ unsigned long long best[BATCH];
  __shared__ int isLast;
  __shared__ int sidx[BATCH];
  int v = blockIdx.x * 256 + threadIdx.x;   // 125*256 == 32000 exactly
  if (threadIdx.x < BATCH) best[threadIdx.x] = 0ull;
  __syncthreads();
  int lane = threadIdx.x & 31;
  float acc = 0.0f;
  for (int b = 0; b < BATCH; ++b) {
    float mx  = fval(rowmax[b]);
    float lse = __logf(rowsum[b]);
    float lp  = logits[(size_t)b * VOCAB + v] - mx - lse;
    float p   = __expf(lp);
    size_t o = ((size_t)b * SEQ + t) * VOCAB + v;
    d_logp[o] = lp;
    d_prob[o] = p;
    acc += p;
    float ue   = u_eps[t * BATCH + b];
    float dnew = (0.05f >= ue) ? UNIFORM_LP : lp;
    float ug   = u_gumbel[((size_t)t * BATCH + b) * VOCAB + v];
    float score = dnew + (-__logf(-__logf(ug)));
    unsigned long long key = (((unsigned long long)fkey(score)) << 32) | (unsigned)v;
#pragma unroll
    for (int m = 16; m >= 1; m >>= 1) {
      unsigned long long o2 = shflxor64(key, m);
      key = (o2 > key) ? o2 : key;
    }
    if (lane == 0) atomicMax(&best[b], key);
  }
  d_mean[(size_t)t * VOCAB + v] = acc * (1.0f / 32.0f);
  __syncthreads();
  if (threadIdx.x < BATCH) atomicMax(argkey + threadIdx.x, best[threadIdx.x]);

  // ---- last-block-done: sample tokens, corrections, next-step input ----
  __threadfence();
  __syncthreads();
  if (threadIdx.x == 0) {
    unsigned prev = atomicAdd(done_ctr, 1u);
    isLast = (prev == gridDim.x - 1) ? 1 : 0;
  }
  __syncthreads();
  if (!isLast) return;
  __threadfence();   // acquire: argkey from all blocks is now final
  if (threadIdx.x < BATCH) {
    int b = threadIdx.x;
    unsigned long long k = argkey[b];
    int s = (int)(unsigned)(k & 0xFFFFFFFFull);
    sidx[b] = s;
    float mx  = fval(rowmax[b]);
    float lse = __logf(rowsum[b]);
    float lp  = logits[(size_t)b * VOCAB + s] - mx - lse;   // recomputed from stable scratch
    float onp = __expf(lp);
    float ue  = u_eps[t * BATCH + b];
    float dnew = (0.05f >= ue) ? UNIFORM_LP : lp;
    float offp = __expf(dnew);
    float con  = fminf(fmaxf(onp, 1e-8f), 1.0f);
    float coff = fminf(fmaxf(offp, 1e-8f), 1.0f);
    out_tok[b * SEQ + t]  = (float)s;
    out_corr[b * SEQ + t] = con / coff;
  }
  __syncthreads();
  for (int i = threadIdx.x; i < BATCH * EMB; i += 256) {
    int b = i >> 7, e = i & (EMB - 1);
    xf16[i] = (_Float16)emb[(size_t)sidx[b] * EMB + e];
  }
}

extern "C" void kernel_launch(void* const* d_in, const int* in_sizes, int n_in,
                              void* d_out, int out_size, void* d_ws, size_t ws_size,
                              hipStream_t stream) {
  const float* embedding = (const float*)d_in[0];
  const float* w_ih      = (const float*)d_in[1];
  const float* w_hh      = (const float*)d_in[2];
  const float* b_ih      = (const float*)d_in[3];
  const float* b_hh      = (const float*)d_in[4];
  const float* w_dist    = (const float*)d_in[5];
  const float* b_dist    = (const float*)d_in[6];
  const float* u_eps     = (const float*)d_in[7];
  const float* u_gumbel  = (const float*)d_in[8];

  float* out = (float*)d_out;
  float* out_tok  = out;                                        // [B,T]
  float* out_corr = out + BATCH * SEQ;                          // [B,T]
  float* d_logp   = out + 2 * BATCH * SEQ;                      // [B,T,V]
  float* d_prob   = d_logp + (size_t)BATCH * SEQ * VOCAB;       // [B,T,V]
  float* d_mean   = d_prob + (size_t)BATCH * SEQ * VOCAB;       // [T,V]

  char* ws = (char*)d_ws;
  auto carve = [&](size_t bytes) { char* p = ws; ws += (bytes + 255) & ~(size_t)255; return p; };
  _Float16* w_dist_h = (_Float16*)carve((size_t)VOCAB * HID * 2);
  _Float16* w_ih_h   = (_Float16*)carve((size_t)3 * HID * EMB * 2);
  _Float16* w_hh_h   = (_Float16*)carve((size_t)3 * HID * HID * 2);
  _Float16* h_f16    = (_Float16*)carve((size_t)BATCH * HID * 2);
  _Float16* x_f16    = (_Float16*)carve((size_t)BATCH * EMB * 2);
  float*    h_f32    = (float*)carve((size_t)BATCH * HID * 4);
  float*    Gi       = (float*)carve((size_t)BATCH * 3 * HID * 4);
  float*    Gh       = (float*)carve((size_t)BATCH * 3 * HID * 4);
  float*    logits   = (float*)carve((size_t)BATCH * VOCAB * 4);
  unsigned* rowmax   = (unsigned*)carve(BATCH * 4);
  float*    rowsum   = (float*)carve(BATCH * 4);
  unsigned long long* argkey = (unsigned long long*)carve(BATCH * 8);
  unsigned* done_ctr = (unsigned*)carve(4);

  // one-time: f32 -> f16 weight conversion + state init (re-run every call; deterministic)
  cvt_f16_kernel<<<(VOCAB * HID + 255) / 256, 256, 0, stream>>>(w_dist_h, w_dist, VOCAB * HID);
  cvt_f16_kernel<<<(3 * HID * EMB + 255) / 256, 256, 0, stream>>>(w_ih_h, w_ih, 3 * HID * EMB);
  cvt_f16_kernel<<<(3 * HID * HID + 255) / 256, 256, 0, stream>>>(w_hh_h, w_hh, 3 * HID * HID);
  init_state_kernel<<<(BATCH * HID + 255) / 256, 256, 0, stream>>>(h_f32, h_f16, x_f16, embedding);

  const int gate_blocks   = (2 * (3 * HID / 16) + 7) / 8;   // 192 waves / 8 = 24
  const int logits_blocks = (VOCAB / 16 + 7) / 8;           // 2000 waves / 8 = 250

  for (int t = 0; t < SEQ; ++t) {
    gates_gemm_kernel<<<gate_blocks, 256, 0, stream>>>(x_f16, w_ih_h, b_ih, Gi,
                                                       h_f16, w_hh_h, b_hh, Gh);
    gru_pointwise_kernel<<<(BATCH * HID + 255) / 256, 256, 0, stream>>>(
        Gi, Gh, h_f32, h_f16, rowmax, rowsum, argkey, done_ctr);
    logits_gemm_kernel<<<logits_blocks, 256, 0, stream>>>(h_f16, w_dist_h, b_dist, logits,
                                                          rowmax);
    row_sumexp_kernel<<<BATCH * 16, 256, 0, stream>>>(logits, rowmax, rowsum);
    finalize_kernel<<<VOCAB / 256, 256, 0, stream>>>(logits, rowmax, rowsum, u_eps, u_gumbel,
                                                     embedding, d_logp, d_prob, d_mean,
                                                     out_tok, out_corr, x_f16, argkey,
                                                     done_ctr, t);
  }
}
